// MultiHeadAttention_56307021250649
// MI455X (gfx1250) — matrix operand order repaired and checked by reading, not compiled, not run on hardware
//
#include <hip/hip_runtime.h>
#include <hip/hip_bf16.h>
#include <math.h>

typedef __bf16 bf16;
typedef __attribute__((ext_vector_type(16))) __bf16 v16bf;
typedef __attribute__((ext_vector_type(8)))  __bf16 v8bf;
typedef __attribute__((ext_vector_type(4)))  __bf16 v4bf;
typedef __attribute__((ext_vector_type(8)))  float  v8f;

#define NFEAT 1024
#define NHEAD 16
#define DK    64
#define BB    4
#define TT    1024
#define ROWS  (BB*TT)   // 4096

union ABu { v16bf v; v8bf h[2]; };

// ---------------------------------------------------------------------------
// f32 -> bf16 elementwise convert, 4 elements/thread (b128 in, b64 out)
// ---------------------------------------------------------------------------
__global__ void cvt_bf16(const float* __restrict__ src, bf16* __restrict__ dst, int n4) {
    int i = blockIdx.x * blockDim.x + threadIdx.x;
    if (i < n4) {
        const float4 f = ((const float4*)src)[i];
        ((v4bf*)dst)[i] = (v4bf){(bf16)f.x, (bf16)f.y, (bf16)f.z, (bf16)f.w};
    }
}

// ---------------------------------------------------------------------------
// 1024x1024 f32 W  ->  bf16 W^T  (LDS 32x32 tile transpose, coalesced both ways)
// ---------------------------------------------------------------------------
__global__ void transpose_bf16(const float* __restrict__ W, bf16* __restrict__ WT) {
    __shared__ float tile[32][33];
    const int bx = blockIdx.x & 31;   // tile col
    const int by = blockIdx.x >> 5;   // tile row
    const int x  = bx * 32 + threadIdx.x;
    const int y0 = by * 32 + threadIdx.y;
    #pragma unroll
    for (int i = 0; i < 32; i += 8)
        tile[threadIdx.y + i][threadIdx.x] = W[(size_t)(y0 + i) * NFEAT + x];
    __syncthreads();
    const int xt = by * 32 + threadIdx.x;   // new col = old row (k)
    const int yt = bx * 32 + threadIdx.y;   // new row = old col (n)
    #pragma unroll
    for (int i = 0; i < 32; i += 8)
        WT[(size_t)(yt + i) * NFEAT + xt] = (bf16)tile[threadIdx.x][threadIdx.y + i];
}

// ---------------------------------------------------------------------------
// GEMM: out = X[4096x1024] @ W + bias via v_wmma_f32_16x16x32_bf16.
// WT is W transposed (bf16, row-major [N][K]) so B-operand loads are contiguous.
// One wave -> 32(M) x 64(N) tile (2x4 C accumulators), K stepped by 32:
// 8 WMMAs per K-step vs 12 b128 loads (B reused across both M-tiles).
// MODE selects the output layout.
// ---------------------------------------------------------------------------
enum { MODE_QK = 0, MODE_V = 1, MODE_F32 = 2 };

template<int MODE, typename OutT>
__global__ void gemm_wmma(const bf16* __restrict__ X, const bf16* __restrict__ WT,
                          const float* __restrict__ bias, OutT* __restrict__ out)
{
    const int lane = threadIdx.x & 31;
    const int wave = threadIdx.x >> 5;
    const int w    = blockIdx.x * (blockDim.x >> 5) + wave;
    const int NT   = NFEAT / 64;    // 16
    const int MT   = ROWS / 32;     // 128
    if (w >= MT * NT) return;
    const int mt = w / NT, nt = w % NT;
    const int lr = lane & 15, half = lane >> 4;

    const bf16* xrow0 = X + (size_t)(mt * 32 + lr) * NFEAT;
    const bf16* xrow1 = xrow0 + (size_t)16 * NFEAT;

    v8f c[2][4];
    #pragma unroll
    for (int mi = 0; mi < 2; ++mi)
        #pragma unroll
        for (int j = 0; j < 4; ++j)
            #pragma unroll
            for (int i = 0; i < 8; ++i) c[mi][j][i] = 0.f;

    for (int k0 = 0; k0 < NFEAT; k0 += 32) {
        ABu a0, a1;
        a0.h[0] = *(const v8bf*)(xrow0 + k0 + half * 16);
        a0.h[1] = *(const v8bf*)(xrow0 + k0 + half * 16 + 8);
        a1.h[0] = *(const v8bf*)(xrow1 + k0 + half * 16);
        a1.h[1] = *(const v8bf*)(xrow1 + k0 + half * 16 + 8);
        #pragma unroll
        for (int j = 0; j < 4; ++j) {
            const bf16* bp = WT + (size_t)(nt * 64 + j * 16 + lr) * NFEAT + k0 + half * 16;
            v16bf b = *(const v16bf*)bp;
            c[0][j] = __builtin_amdgcn_wmma_f32_16x16x32_bf16(
                          false, a0.v, false, b, (short)0, c[0][j], false, false);
            c[1][j] = __builtin_amdgcn_wmma_f32_16x16x32_bf16(
                          false, a1.v, false, b, (short)0, c[1][j], false, false);
        }
    }

    #pragma unroll
    for (int mi = 0; mi < 2; ++mi) {
        #pragma unroll
        for (int j = 0; j < 4; ++j) {
            const int   col = nt * 64 + j * 16 + lr;
            const float bv  = bias[col];
            #pragma unroll
            for (int r = 0; r < 8; ++r) {
                const int g   = mt * 32 + mi * 16 + half * 8 + r;   // global row
                const float v = c[mi][j][r] + bv;
                if constexpr (MODE == MODE_F32) {
                    out[(size_t)g * NFEAT + col] = v;
                } else {
                    const int b = g >> 10, t = g & 1023, h = col >> 6, d = col & 63;
                    if constexpr (MODE == MODE_QK)       // [B,H,T,D]
                        out[(((size_t)(b * NHEAD + h) * TT) + t) * DK + d] = (bf16)v;
                    else                                  // MODE_V: [B,H,D,T]
                        out[(((size_t)(b * NHEAD + h) * DK) + d) * TT + t] = (bf16)v;
                }
            }
        }
    }
}

// ---------------------------------------------------------------------------
// Flash-style attention: one wave per (b, h, 16-query tile).
// Q,K: [B,H,T,D] bf16.  Vt: [B,H,D,T] bf16 (transposed).  mask: bool bytes.
// Online softmax over 32-key chunks; P tile goes C-layout -> A-layout via LDS.
// Output: attn_out [B*T, H*D] bf16 (heads re-concatenated, row-major).
// ---------------------------------------------------------------------------
__global__ void attn_kernel(const bf16* __restrict__ Q, const bf16* __restrict__ K,
                            const bf16* __restrict__ Vt, const unsigned char* __restrict__ mask,
                            bf16* __restrict__ out)
{
    __shared__ alignas(32) bf16 pa[16][40];   // 80B row stride: 16B aligned, conflict-free
    const int lane = threadIdx.x & 31;
    const int lr = lane & 15, half = lane >> 4;
    const int bid = blockIdx.x;
    const int qt = bid & 63;
    const int h  = (bid >> 6) & 15;
    const int b  = bid >> 10;
    const int q0 = qt * 16;

    const bf16* Qh = Q  + (size_t)(b * NHEAD + h) * TT * DK;
    const bf16* Kh = K  + (size_t)(b * NHEAD + h) * TT * DK;
    const bf16* Vh = Vt + (size_t)(b * NHEAD + h) * DK * TT;
    const unsigned char* Mh = mask + (size_t)b * TT * TT;

    // Q A-operand tiles for d=0..31 and d=32..63 (loaded once)
    ABu aq[2];
    {
        const bf16* qrow = Qh + (size_t)(q0 + lr) * DK;
        #pragma unroll
        for (int s = 0; s < 2; ++s) {
            aq[s].h[0] = *(const v8bf*)(qrow + s * 32 + half * 16);
            aq[s].h[1] = *(const v8bf*)(qrow + s * 32 + half * 16 + 8);
        }
    }

    v8f o[4];
    #pragma unroll
    for (int j = 0; j < 4; ++j)
        #pragma unroll
        for (int i = 0; i < 8; ++i) o[j][i] = 0.f;
    float m[8], l[8];
    #pragma unroll
    for (int r = 0; r < 8; ++r) { m[r] = -INFINITY; l[r] = 0.f; }

    const float scale = 0.125f;   // 1/sqrt(64)

    for (int kt = 0; kt < TT; kt += 32) {
        // scores: two 16-key C tiles, each = 2 WMMAs over d
        v8f s[2];
        #pragma unroll
        for (int nt = 0; nt < 2; ++nt) {
            const int key = kt + nt * 16 + lr;
            v16bf bk0 = *(const v16bf*)(Kh + (size_t)key * DK + half * 16);
            v16bf bk1 = *(const v16bf*)(Kh + (size_t)key * DK + 32 + half * 16);
            v8f acc;
            #pragma unroll
            for (int i = 0; i < 8; ++i) acc[i] = 0.f;
            acc = __builtin_amdgcn_wmma_f32_16x16x32_bf16(
                      false, aq[0].v, false, bk0, (short)0, acc, false, false);
            acc = __builtin_amdgcn_wmma_f32_16x16x32_bf16(
                      false, aq[1].v, false, bk1, (short)0, acc, false, false);
            s[nt] = acc;
        }

        // scale, mask, per-lane row max
        unsigned mb0 = 0, mb1 = 0;
        float mx[8];
        #pragma unroll
        for (int r = 0; r < 8; ++r) {
            const int row = q0 + half * 8 + r;
            float v0 = s[0][r] * scale;
            float v1 = s[1][r] * scale;
            if (Mh[(size_t)row * TT + kt + lr])      { v0 = -1e30f; mb0 |= (1u << r); }
            if (Mh[(size_t)row * TT + kt + 16 + lr]) { v1 = -1e30f; mb1 |= (1u << r); }
            s[0][r] = v0; s[1][r] = v1;
            mx[r] = fmaxf(v0, v1);
        }
        // reduce row max across the 16-lane half (wave32 shfl)
        #pragma unroll
        for (int r = 0; r < 8; ++r) {
            float v = mx[r];
            for (int off = 1; off < 16; off <<= 1)
                v = fmaxf(v, __shfl_xor(v, off));
            mx[r] = v;
        }

        // online softmax update; stage P into LDS (C-layout write)
        float ps[8];
        #pragma unroll
        for (int r = 0; r < 8; ++r) {
            const float mn    = fmaxf(m[r], mx[r]);
            const float alpha = __expf(m[r] - mn);
            m[r] = mn;
            const float p0 = ((mb0 >> r) & 1) ? 0.f : __expf(s[0][r] - mn);
            const float p1 = ((mb1 >> r) & 1) ? 0.f : __expf(s[1][r] - mn);
            ps[r] = p0 + p1;
            l[r] *= alpha;
            pa[half * 8 + r][lr]      = (bf16)p0;
            pa[half * 8 + r][16 + lr] = (bf16)p1;
            #pragma unroll
            for (int j = 0; j < 4; ++j) o[j][r] *= alpha;
        }
        #pragma unroll
        for (int r = 0; r < 8; ++r) {
            float v = ps[r];
            for (int off = 1; off < 16; off <<= 1)
                v += __shfl_xor(v, off);
            l[r] += v;
        }
        __syncthreads();

        // read P back in A-layout
        ABu ap;
        ap.h[0] = *(const v8bf*)(&pa[lr][half * 16]);
        ap.h[1] = *(const v8bf*)(&pa[lr][half * 16 + 8]);

        // O += P @ V  (4 d-tiles)
        #pragma unroll
        for (int j = 0; j < 4; ++j) {
            v16bf bv = *(const v16bf*)(Vh + (size_t)(j * 16 + lr) * TT + kt + half * 16);
            o[j] = __builtin_amdgcn_wmma_f32_16x16x32_bf16(
                       false, ap.v, false, bv, (short)0, o[j], false, false);
        }
        __syncthreads();
    }

    // normalize and store heads re-concatenated: [B*T, H*D]
    #pragma unroll
    for (int j = 0; j < 4; ++j) {
        const int col = h * DK + j * 16 + lr;
        #pragma unroll
        for (int r = 0; r < 8; ++r) {
            const int grow = b * TT + q0 + half * 8 + r;
            const float inv = (l[r] > 0.f) ? (1.f / l[r]) : 0.f;
            out[(size_t)grow * NFEAT + col] = (bf16)(o[j][r] * inv);
        }
    }
}

// ---------------------------------------------------------------------------
// Launch
// ---------------------------------------------------------------------------
extern "C" void kernel_launch(void* const* d_in, const int* in_sizes, int n_in,
                              void* d_out, int out_size, void* d_ws, size_t ws_size,
                              hipStream_t stream)
{
    (void)in_sizes; (void)n_in; (void)out_size; (void)ws_size;
    const float* query = (const float*)d_in[0];
    const float* key   = (const float*)d_in[1];
    const float* value = (const float*)d_in[2];
    const unsigned char* mask = (const unsigned char*)d_in[3];
    const float* Wq = (const float*)d_in[4];
    const float* bq = (const float*)d_in[5];
    const float* Wk = (const float*)d_in[6];
    const float* bk = (const float*)d_in[7];
    const float* Wv = (const float*)d_in[8];
    const float* bv = (const float*)d_in[9];
    const float* Wo = (const float*)d_in[10];
    const float* bo = (const float*)d_in[11];
    float* out = (float*)d_out;

    char* ws = (char*)d_ws;
    const size_t SZ_ACT = (size_t)ROWS * NFEAT * sizeof(bf16);   // 8 MiB
    const size_t SZ_W   = (size_t)NFEAT * NFEAT * sizeof(bf16);  // 2 MiB
    bf16* xq   = (bf16*)ws; ws += SZ_ACT;
    bf16* xk   = (bf16*)ws; ws += SZ_ACT;
    bf16* xv   = (bf16*)ws; ws += SZ_ACT;
    bf16* WTq  = (bf16*)ws; ws += SZ_W;
    bf16* WTk  = (bf16*)ws; ws += SZ_W;
    bf16* WTv  = (bf16*)ws; ws += SZ_W;
    bf16* WTo  = (bf16*)ws; ws += SZ_W;
    bf16* qb   = (bf16*)ws; ws += SZ_ACT;
    bf16* kb   = (bf16*)ws; ws += SZ_ACT;
    bf16* vb   = (bf16*)ws; ws += SZ_ACT;
    bf16* attn = (bf16*)ws; ws += SZ_ACT;   // total 64 MiB

    const int n4 = (ROWS * NFEAT) / 4;
    cvt_bf16<<<(n4 + 255) / 256, 256, 0, stream>>>(query, xq, n4);
    cvt_bf16<<<(n4 + 255) / 256, 256, 0, stream>>>(key,   xk, n4);
    cvt_bf16<<<(n4 + 255) / 256, 256, 0, stream>>>(value, xv, n4);

    dim3 tb(32, 8);
    transpose_bf16<<<1024, tb, 0, stream>>>(Wq, WTq);
    transpose_bf16<<<1024, tb, 0, stream>>>(Wk, WTk);
    transpose_bf16<<<1024, tb, 0, stream>>>(Wv, WTv);
    transpose_bf16<<<1024, tb, 0, stream>>>(Wo, WTo);

    const int waves  = (ROWS / 32) * (NFEAT / 64);  // 2048
    const int blocks = waves / 8;                   // 256 threads = 8 waves/block
    gemm_wmma<MODE_QK, bf16><<<blocks, 256, 0, stream>>>(xq, WTq, bq, qb);
    gemm_wmma<MODE_QK, bf16><<<blocks, 256, 0, stream>>>(xk, WTk, bk, kb);
    gemm_wmma<MODE_V,  bf16><<<blocks, 256, 0, stream>>>(xv, WTv, bv, vb);

    attn_kernel<<<BB * NHEAD * (TT / 16), 32, 0, stream>>>(qb, kb, vb, mask, attn);

    gemm_wmma<MODE_F32, float><<<blocks, 256, 0, stream>>>(attn, WTo, bo, out);
}
